// MyRNN_58325655879943
// MI455X (gfx1250) — matrix operand order repaired and checked by reading, not compiled, not run on hardware
//
#include <hip/hip_runtime.h>
#include <hip/hip_bf16.h>

#define RNN_VOCAB 10000
#define RNN_EMB   100
#define RNN_EMBP  128   // EMB zero-padded to multiple of 32 (WMMA K granularity)
#define RNN_SEQ   80
#define RNN_UNITS 512
#define RNN_BATCH 1024

typedef __attribute__((ext_vector_type(16))) __bf16 bf16x16;
typedef __attribute__((ext_vector_type(8)))  float  floatx8;

// ---------------------------------------------------------------------------
// Hardware tanh (CDNA5 V_TANH_F32 transcendental). Fallback inline asm keeps
// the mandatory 1-independent-op spacing after a TRANS op (ISA 15.5 §7.4.3).
// ---------------------------------------------------------------------------
__device__ __forceinline__ float hw_tanh(float x) {
#if __has_builtin(__builtin_amdgcn_tanhf)
    return __builtin_amdgcn_tanhf(x);
#else
    float y;
    asm volatile("v_tanh_f32 %0, %1\n\tv_nop" : "=v"(y) : "v"(x));
    return y;
#endif
}

// ---------------------------------------------------------------------------
// Prep: convert fp32 weight [K][N] -> bf16 transposed [N][Kpad] (zero padded).
// ---------------------------------------------------------------------------
__global__ void wt_prep_kernel(const float* __restrict__ src,
                               __bf16* __restrict__ dst,
                               int K, int N, int Kpad) {
    int idx = blockIdx.x * blockDim.x + threadIdx.x;
    int total = N * Kpad;
    if (idx >= total) return;
    int n = idx / Kpad;
    int k = idx - n * Kpad;
    float v = (k < K) ? src[(size_t)k * N + n] : 0.0f;
    dst[idx] = (__bf16)v;
}

// ---------------------------------------------------------------------------
// Fragment loaders matching CDNA5 ISA 7.12.2 wave32 VGPR layouts.
// ---------------------------------------------------------------------------

// A (16x32 bf16, M x K) from row-major LDS buffer with row stride `stride`.
__device__ __forceinline__ bf16x16 load_a_frag(const __bf16* base, int stride,
                                               int k0, int mrow, int hi) {
    const __bf16* p = base + mrow * stride + k0 + hi * 8;
    union { bf16x16 v; float4 q[2]; } u;
    u.q[0] = *(const float4*)(p);
    u.q[1] = *(const float4*)(p + 16);
    return u.v;
}

// B (32x16 bf16, K x N) from bf16 transposed weights Wt[N][Kw] in global.
__device__ __forceinline__ bf16x16 load_b_frag(const __bf16* __restrict__ wt,
                                               int kw, int n0, int k0,
                                               int mrow, int hi) {
    const __bf16* p = wt + (size_t)(n0 + mrow) * kw + k0 + hi * 8;
    union { bf16x16 v; float4 q[2]; } u;
    u.q[0] = *(const float4*)(p);
    u.q[1] = *(const float4*)(p + 16);
    return u.v;
}

// ---------------------------------------------------------------------------
// Persistent RNN kernel: one workgroup owns a 16-row batch tile for all 80
// steps. h0/h1 live in LDS as bf16 [16][512]. 8 waves x 64 columns = 512.
// ---------------------------------------------------------------------------
__launch_bounds__(256)
__global__ void rnn_wmma_kernel(const int*   __restrict__ tokens,   // [B][SEQ]
                                const float* __restrict__ emb,      // [VOCAB][EMB]
                                const __bf16* __restrict__ wx0t,    // [512][128]
                                const __bf16* __restrict__ wh0t,    // [512][512]
                                const __bf16* __restrict__ wx1t,    // [512][512]
                                const __bf16* __restrict__ wh1t,    // [512][512]
                                const float* __restrict__ b0,
                                const float* __restrict__ b1,
                                const float* __restrict__ wout,     // [512]
                                const float* __restrict__ bout,     // [1]
                                float*       __restrict__ out) {    // [B]
    __shared__ __bf16 h0[16 * RNN_UNITS];
    __shared__ __bf16 h1[16 * RNN_UNITS];
    __shared__ __bf16 xb[16 * RNN_EMBP];
    __shared__ float  red[16 * 16];

    const int tid   = threadIdx.x;
    const int lane  = tid & 31;
    const int wave  = tid >> 5;          // 0..7
    const int mrow  = lane & 15;
    const int hi    = lane >> 4;
    const int n0    = wave * 64;         // this wave's 64-column slab
    const int bbase = blockIdx.x * 16;   // batch tile start

    // zero-init recurrent state AND xb (pad cols 100..127 stay zero forever)
    for (int i = tid; i < 16 * RNN_UNITS; i += 256) {
        h0[i] = (__bf16)0.0f;
        h1[i] = (__bf16)0.0f;
    }
    for (int i = tid; i < 16 * RNN_EMBP; i += 256) xb[i] = (__bf16)0.0f;
    __syncthreads();

    for (int t = 0; t < RNN_SEQ; ++t) {
        // ---- stage x_t tile: vector gather of embedding rows ----
        {
            int row = tid >> 4;                 // 0..15
            int seg = tid & 15;                 // 8-elem segment
            int tok = tokens[(bbase + row) * RNN_SEQ + t];
            const float* er = emb + (size_t)tok * RNN_EMB;
            if (seg < 12) {                     // cols 0..95: full 8-elem copy
                float4 f0 = *(const float4*)(er + seg * 8);
                float4 f1 = *(const float4*)(er + seg * 8 + 4);
                union { __bf16 h[8]; float4 q; } u;
                u.h[0] = (__bf16)f0.x; u.h[1] = (__bf16)f0.y;
                u.h[2] = (__bf16)f0.z; u.h[3] = (__bf16)f0.w;
                u.h[4] = (__bf16)f1.x; u.h[5] = (__bf16)f1.y;
                u.h[6] = (__bf16)f1.z; u.h[7] = (__bf16)f1.w;
                *(float4*)&xb[row * RNN_EMBP + seg * 8] = u.q;
            } else if (seg == 12) {             // cols 96..99
                float4 f0 = *(const float4*)(er + 96);
                union { __bf16 h[4]; float2 q; } u;
                u.h[0] = (__bf16)f0.x; u.h[1] = (__bf16)f0.y;
                u.h[2] = (__bf16)f0.z; u.h[3] = (__bf16)f0.w;
                *(float2*)&xb[row * RNN_EMBP + 96] = u.q;
            }
        }
        __syncthreads();   // xb ready; prev-step h1 writes also fenced

        // ---- layer 0: acc = x_t @ Wx0 + h0 @ Wh0 ----
        floatx8 acc[4] = {};
        #pragma unroll
        for (int k = 0; k < RNN_EMBP; k += 32) {
            bf16x16 a = load_a_frag(xb, RNN_EMBP, k, mrow, hi);
            #pragma unroll
            for (int ti = 0; ti < 4; ++ti) {
                bf16x16 b = load_b_frag(wx0t, RNN_EMBP, n0 + ti * 16, k, mrow, hi);
                acc[ti] = __builtin_amdgcn_wmma_f32_16x16x32_bf16(
                    false, a, false, b, (short)0, acc[ti], false, false);
            }
        }
        #pragma unroll 2
        for (int k = 0; k < RNN_UNITS; k += 32) {
            bf16x16 a = load_a_frag(h0, RNN_UNITS, k, mrow, hi);
            #pragma unroll
            for (int ti = 0; ti < 4; ++ti) {
                bf16x16 b = load_b_frag(wh0t, RNN_UNITS, n0 + ti * 16, k, mrow, hi);
                acc[ti] = __builtin_amdgcn_wmma_f32_16x16x32_bf16(
                    false, a, false, b, (short)0, acc[ti], false, false);
            }
        }
        __syncthreads();   // every wave done READING old h0 / xb

        // ---- bias + hw tanh + store new h0 (C/D: M = r + hi*8, N = mrow) ----
        #pragma unroll
        for (int ti = 0; ti < 4; ++ti) {
            int n = n0 + ti * 16 + mrow;
            float bv = b0[n];
            #pragma unroll
            for (int r = 0; r < 8; ++r) {
                int m = r + hi * 8;
                h0[m * RNN_UNITS + n] = (__bf16)hw_tanh(acc[ti][r] + bv);
            }
        }
        __syncthreads();   // new h0 visible

        // ---- layer 1: acc = h0_new @ Wx1 + h1 @ Wh1 ----
        floatx8 acc2[4] = {};
        #pragma unroll 2
        for (int k = 0; k < RNN_UNITS; k += 32) {
            bf16x16 a = load_a_frag(h0, RNN_UNITS, k, mrow, hi);
            #pragma unroll
            for (int ti = 0; ti < 4; ++ti) {
                bf16x16 b = load_b_frag(wx1t, RNN_UNITS, n0 + ti * 16, k, mrow, hi);
                acc2[ti] = __builtin_amdgcn_wmma_f32_16x16x32_bf16(
                    false, a, false, b, (short)0, acc2[ti], false, false);
            }
        }
        #pragma unroll 2
        for (int k = 0; k < RNN_UNITS; k += 32) {
            bf16x16 a = load_a_frag(h1, RNN_UNITS, k, mrow, hi);
            #pragma unroll
            for (int ti = 0; ti < 4; ++ti) {
                bf16x16 b = load_b_frag(wh1t, RNN_UNITS, n0 + ti * 16, k, mrow, hi);
                acc2[ti] = __builtin_amdgcn_wmma_f32_16x16x32_bf16(
                    false, a, false, b, (short)0, acc2[ti], false, false);
            }
        }
        __syncthreads();   // every wave done READING old h1

        #pragma unroll
        for (int ti = 0; ti < 4; ++ti) {
            int n = n0 + ti * 16 + mrow;
            float bv = b1[n];
            #pragma unroll
            for (int r = 0; r < 8; ++r) {
                int m = r + hi * 8;
                h1[m * RNN_UNITS + n] = (__bf16)hw_tanh(acc2[ti][r] + bv);
            }
        }
        // next iteration's first __syncthreads() fences these writes
    }
    __syncthreads();

    // ---- logits = h1 @ Wout + bout ; sigmoid ----
    {
        int row = tid >> 4;       // 0..15
        int seg = tid & 15;       // 16 segments x 32 elems
        float s = 0.0f;
        #pragma unroll 4
        for (int j = seg * 32; j < seg * 32 + 32; ++j)
            s += (float)h1[row * RNN_UNITS + j] * wout[j];
        red[row * 16 + seg] = s;
    }
    __syncthreads();
    if ((tid & 15) == 0) {
        int row = tid >> 4;
        float s = bout[0];
        #pragma unroll
        for (int j = 0; j < 16; ++j) s += red[row * 16 + j];
        out[bbase + row] = 1.0f / (1.0f + __expf(-s));
    }
}

// ---------------------------------------------------------------------------
// Launch
// ---------------------------------------------------------------------------
extern "C" void kernel_launch(void* const* d_in, const int* in_sizes, int n_in,
                              void* d_out, int out_size, void* d_ws, size_t ws_size,
                              hipStream_t stream) {
    const int*   tokens = (const int*)  d_in[0];
    const float* emb    = (const float*)d_in[1];
    const float* Wx0    = (const float*)d_in[2];
    const float* Wh0    = (const float*)d_in[3];
    const float* b0     = (const float*)d_in[4];
    const float* Wx1    = (const float*)d_in[5];
    const float* Wh1    = (const float*)d_in[6];
    const float* b1     = (const float*)d_in[7];
    const float* Wout   = (const float*)d_in[8];
    const float* bout   = (const float*)d_in[9];
    float* out = (float*)d_out;

    // workspace layout: bf16 transposed weights
    char* ws = (char*)d_ws;
    size_t off = 0;
    __bf16* wx0t = (__bf16*)(ws + off); off += (size_t)RNN_UNITS * RNN_EMBP  * 2;
    __bf16* wh0t = (__bf16*)(ws + off); off += (size_t)RNN_UNITS * RNN_UNITS * 2;
    __bf16* wx1t = (__bf16*)(ws + off); off += (size_t)RNN_UNITS * RNN_UNITS * 2;
    __bf16* wh1t = (__bf16*)(ws + off); off += (size_t)RNN_UNITS * RNN_UNITS * 2;
    (void)ws_size; (void)in_sizes; (void)n_in; (void)out_size;

    int tot;
    tot = RNN_UNITS * RNN_EMBP;
    wt_prep_kernel<<<(tot + 255) / 256, 256, 0, stream>>>(Wx0, wx0t, RNN_EMB, RNN_UNITS, RNN_EMBP);
    tot = RNN_UNITS * RNN_UNITS;
    wt_prep_kernel<<<(tot + 255) / 256, 256, 0, stream>>>(Wh0, wh0t, RNN_UNITS, RNN_UNITS, RNN_UNITS);
    wt_prep_kernel<<<(tot + 255) / 256, 256, 0, stream>>>(Wx1, wx1t, RNN_UNITS, RNN_UNITS, RNN_UNITS);
    wt_prep_kernel<<<(tot + 255) / 256, 256, 0, stream>>>(Wh1, wh1t, RNN_UNITS, RNN_UNITS, RNN_UNITS);

    rnn_wmma_kernel<<<RNN_BATCH / 16, 256, 0, stream>>>(
        tokens, emb, wx0t, wh0t, wx1t, wh1t, b0, b1, Wout, bout, out);
}
